// Net_111669150344
// MI455X (gfx1250) — matrix operand order repaired and checked by reading, not compiled, not run on hardware
//
#include <hip/hip_runtime.h>
#include <hip/hip_bf16.h>

typedef __attribute__((ext_vector_type(16))) _Float16 v16h;
typedef __attribute__((ext_vector_type(8)))  _Float16 v8h;
typedef __attribute__((ext_vector_type(8)))  float    v8f;

#define TPB 256
#define WV  8          // waves per block in k_mlp (wave32 -> 256 threads)

__device__ __forceinline__ float sigmoidf_(float x){ return 1.0f/(1.0f+expf(-x)); }

__device__ __forceinline__ void atomicMaxF(float* a, float v){
  if (v >= 0.0f) atomicMax((int*)a, __float_as_int(v));
  else           atomicMin((unsigned int*)a, __float_as_uint(v));
}

// ---------------------------------------------------------------------------
// Fused up-to-3-layer MLP:  Y = act(...relu(X@W0+b0)@W1+b1...)
// Compile-time shapes -> fully unrolled, static v_wmma_f32_16x16x32_f16 chain.
// VGPR layouts per CDNA5 ISA 7.12.2:
//   A: lane l -> row l&15 ; elem j -> K = (j<8 ? 8*(l>>4)+j : 16+8*(l>>4)+j-8)
//   B: lane l -> col l&15 ; elem j -> K = (j<8 ? 8*(l>>4)+j : 16+8*(l>>4)+j-8)      (staged transposed)
//   C: lane l -> col l&15 ; vgpr v -> row v + 8*(l>>4)
// Weights staged transposed+padded in LDS; activations staged per-wave in LDS.
// ---------------------------------------------------------------------------
template<int KIN> __device__ __forceinline__
void stageW(const float* __restrict__ W, int NOUT_rt, int NOUT, _Float16* __restrict__ Wt){
  // transposed, padded: Wt[n][k], n < NP=ceil(NOUT/16)*16, k < KP=ceil(KIN/32)*32
  (void)NOUT_rt;
  const int KP = ((KIN + 31)/32)*32;
  const int NP = ((NOUT + 15)/16)*16;
  for (int i = threadIdx.x; i < NP*KP; i += TPB){
    int n = i / KP, k = i % KP;
    float v = (n < NOUT && k < KIN) ? W[(size_t)k*NOUT + n] : 0.0f;
    Wt[i] = (_Float16)v;
  }
}
__device__ __forceinline__
void stageB(const float* __restrict__ B, int NOUT, float* __restrict__ bs){
  for (int i = threadIdx.x; i < 64; i += TPB)
    bs[i] = (B != nullptr && i < NOUT) ? B[i] : 0.0f;
}

template<int KIN, int NOUT, bool FIRST, bool LAST>
__device__ __forceinline__ void mlp_layer(
    const float* __restrict__ X, int M, int lda, int row0,
    const _Float16* __restrict__ Wt,      // LDS, transposed [NP][KP]
    const float* __restrict__ bs,         // LDS, padded bias [64]
    const _Float16* __restrict__ sIn,     // per-wave LDS act in  [16][64]
    _Float16* __restrict__ sOut,          // per-wave LDS act out [16][64]
    float* __restrict__ Y, int act_last, int l15, int l16)
{
  constexpr int KT = (KIN + 31)/32;
  constexpr int NT = (NOUT + 15)/16;
  constexpr int KP = KT*32;

  v16h af[KT];
  if (FIRST) {
    // KIN <= 32 for every first layer in this net -> KT == 1
    int r = row0 + l15; if (r >= M) r = M - 1;            // clamp; tail discarded at store
    const float* xr = X + (size_t)r * lda;
    float tmp[KIN];
    #pragma unroll
    for (int k = 0; k < KIN; ++k) tmp[k] = xr[k];         // unconditional, contiguous
    #pragma unroll
    for (int t = 0; t < KT; ++t) {
      v16h a;
      #pragma unroll
      for (int j = 0; j < 16; ++j) {
        int kb = t*32 + ((j < 8) ? j : (16 + (j - 8)));   // l16==0 index
        int k0 = kb, k1 = kb + 8;                          // l16==1 index
        float v0 = (k0 < KIN) ? tmp[k0] : 0.0f;            // compile-time folds
        float v1 = (k1 < KIN) ? tmp[k1] : 0.0f;
        a[j] = (_Float16)(l16 ? v1 : v0);
      }
      af[t] = a;
    }
  } else {
    #pragma unroll
    for (int t = 0; t < KT; ++t) {
      const _Float16* sp = sIn + l15*64 + t*32 + l16*8;    // 16B-aligned
      v8h lo = *(const v8h*)sp;
      v8h hi = *(const v8h*)(sp + 16);
      v16h a;
      #pragma unroll
      for (int j = 0; j < 8; ++j){ a[j] = lo[j]; a[8+j] = hi[j]; }
      af[t] = a;
    }
  }

  #pragma unroll
  for (int t2 = 0; t2 < NT; ++t2) {
    const int nn = t2*16 + l15;
    v8f acc = {};
    #pragma unroll
    for (int t = 0; t < KT; ++t) {
      const _Float16* wp = Wt + (size_t)nn*KP + t*32 + l16*8;
      v8h blo = *(const v8h*)wp;
      v8h bhi = *(const v8h*)(wp + 16);
      v16h b;
      #pragma unroll
      for (int j = 0; j < 8; ++j){ b[j] = blo[j]; b[8+j] = bhi[j]; }
      acc = __builtin_amdgcn_wmma_f32_16x16x32_f16(
              false, af[t], false, b, (short)0, acc, false, false);
    }
    const float bias = bs[nn];
    #pragma unroll
    for (int v = 0; v < 8; ++v) {
      float val = acc[v] + bias;
      const int mr = v + l16*8;
      if (!LAST) {
        val = fmaxf(val, 0.0f);                            // hidden: ReLU
        sOut[mr*64 + nn] = (_Float16)((nn < NOUT) ? val : 0.0f);
      } else {
        const int r = row0 + mr;
        if (r < M && nn < NOUT) {
          if      (act_last == 1) val = fmaxf(val, 0.0f);
          else if (act_last == 2) val = sigmoidf_(val);
          Y[(size_t)r * NOUT + nn] = val;
        }
      }
    }
  }
}

template<int D0,int D1,int D2,int D3,int NL>
__global__ __launch_bounds__(256) void k_mlp(
    const float* __restrict__ X, int M, int lda,
    const float* __restrict__ W0, const float* __restrict__ B0,
    const float* __restrict__ W1, const float* __restrict__ B1,
    const float* __restrict__ W2, const float* __restrict__ B2,
    int act_last, float* __restrict__ Y)
{
  constexpr int S0 = (((D1+15)/16)*16) * (((D0+31)/32)*32);
  constexpr int S1 = (NL > 1) ? (((D2+15)/16)*16) * (((D1+31)/32)*32) : 0;
  constexpr int S2 = (NL > 2) ? (((D3+15)/16)*16) * (((D2+31)/32)*32) : 0;
  __shared__ alignas(64) _Float16 wts[S0 + S1 + S2];
  __shared__ alignas(64) _Float16 act[(NL > 1) ? (WV*16*64) : WV];
  __shared__ float bstage[3][64];

  stageW<D0>(W0, D1, D1, wts);
  stageB(B0, D1, bstage[0]);
  if constexpr (NL > 1) { stageW<D1>(W1, D2, D2, wts + S0); stageB(B1, D2, bstage[1]); }
  if constexpr (NL > 2) { stageW<D2>(W2, D3, D3, wts + S0 + S1); stageB(B2, D3, bstage[2]); }
  __syncthreads();

  const int wave = threadIdx.x >> 5;
  const int lane = threadIdx.x & 31;
  const int l15  = lane & 15;
  const int l16  = lane >> 4;
  const int row0 = (blockIdx.x * WV + wave) * 16;
  _Float16* myact = &act[(NL > 1) ? (wave*16*64) : 0];

  mlp_layer<D0, D1, true, (NL==1)>(X, M, lda, row0, wts, bstage[0],
                                   myact, myact, Y, act_last, l15, l16);
  if constexpr (NL > 1)
    mlp_layer<D1, D2, false, (NL==2)>(X, M, lda, row0, wts + S0, bstage[1],
                                      myact, myact, Y, act_last, l15, l16);
  if constexpr (NL > 2)
    mlp_layer<D2, D3, false, true>(X, M, lda, row0, wts + S0 + S1, bstage[2],
                                   myact, myact, Y, act_last, l15, l16);
}

// --------------------------- scatter / utility kernels ----------------------
__global__ void k_fill(float* p, float v, int n){
  int i = blockIdx.x*blockDim.x + threadIdx.x;
  if (i < n) p[i] = v;
}
__global__ void k_rowbc(float* out, const float* bias, int n, int c){
  int i = blockIdx.x*blockDim.x + threadIdx.x;
  if (i < n*c) out[i] = bias[i % c];
}
// out[e] = { r[src]-r[dst] (3), tail[dst, toff : toff+tw) }
__global__ void k_edge_cat(const float* __restrict__ r, const int* __restrict__ ei, int E_,
                           const float* __restrict__ tail, int tstride, int toff, int tw,
                           float* __restrict__ out){
  int e = blockIdx.x*blockDim.x + threadIdx.x;
  if (e >= E_) return;
  int s = ei[e], d = ei[E_ + e];
  int w = 3 + tw;
  float* o = out + (size_t)e * w;
  for (int c = 0; c < 3; ++c)  o[c]   = r[s*3+c] - r[d*3+c];
  for (int c = 0; c < tw; ++c) o[3+c] = tail[(size_t)d*tstride + toff + c];
}
__global__ void k_gat_logits(const float* __restrict__ h, const float* __restrict__ eh,
                             const float* __restrict__ as, const float* __restrict__ ad,
                             const float* __restrict__ ae, const int* __restrict__ ei,
                             int E_, int o, float* __restrict__ eo){
  int e = blockIdx.x*blockDim.x + threadIdx.x;
  if (e >= E_) return;
  int s = ei[e], d = ei[E_ + e];
  float acc = 0.f;
  for (int c = 0; c < o; ++c)
    acc += h[(size_t)s*o+c]*as[c] + h[(size_t)d*o+c]*ad[c] + eh[(size_t)e*o+c]*ae[c];
  eo[e] = (acc > 0.f) ? acc : 0.2f*acc;   // leaky_relu(0.2)
}
__global__ void k_segmax(const float* __restrict__ eo, const int* __restrict__ ei, int E_, float* m){
  int e = blockIdx.x*blockDim.x + threadIdx.x;
  if (e >= E_) return;
  atomicMaxF(&m[ei[E_ + e]], eo[e]);
}
__global__ void k_gat_exp(const float* eo, const float* m, const int* ei, int E_,
                          float* ex, float* den){
  int e = blockIdx.x*blockDim.x + threadIdx.x;
  if (e >= E_) return;
  int d = ei[E_ + e];
  float v = expf(eo[e] - m[d]);
  ex[e] = v;
  atomicAdd(&den[d], v);
}
__global__ void k_gat_agg(const float* ex, const float* den, const float* h,
                          const int* ei, int E_, int o, float* out){
  int e = blockIdx.x*blockDim.x + threadIdx.x;
  if (e >= E_) return;
  int s = ei[e], d = ei[E_ + e];
  float a = ex[e] / (den[d] + 1e-16f);
  for (int c = 0; c < o; ++c)
    atomicAdd(&out[(size_t)d*o+c], a * h[(size_t)s*o+c]);
}
__global__ void k_colstats(const float* x, int n, int c, float* stat){
  int i = blockIdx.x*blockDim.x + threadIdx.x;
  if (i >= n) return;
  for (int j = 0; j < c; ++j){
    float v = x[(size_t)i*c+j];
    atomicAdd(&stat[j], v);
    atomicAdd(&stat[64+j], v*v);
  }
}
__global__ void k_bn(const float* x, const float* stat, const float* g, const float* b,
                     int n, int c, float* y){
  int i = blockIdx.x*blockDim.x + threadIdx.x;
  if (i >= n*c) return;
  int j = i % c;
  float mu  = stat[j] / (float)n;
  float var = stat[64+j] / (float)n - mu*mu;
  y[i] = (x[i]-mu) * rsqrtf(var + 1e-5f) * g[j] + b[j];
}
__global__ void k_colsum6(const float* x, const float* xp, int n, float* stat){
  int i = blockIdx.x*blockDim.x + threadIdx.x;
  if (i >= n) return;
  for (int k = 0; k < 3; ++k){
    atomicAdd(&stat[k],   x[(size_t)i*7+k]);
    atomicAdd(&stat[3+k], xp[(size_t)i*7+k]);
  }
}
#define KNN_TILE 1024
__global__ __launch_bounds__(256) void k_knn(const float* __restrict__ x,
                                             const float* __restrict__ xp,
                                             const float* __restrict__ stat,
                                             int n, int* __restrict__ idx){
  __shared__ float sp[KNN_TILE*3];
  int i = blockIdx.x*blockDim.x + threadIdx.x;
  float inv = 1.0f / (float)n;
  float ox = (stat[0]-stat[3])*inv, oy = (stat[1]-stat[4])*inv, oz = (stat[2]-stat[5])*inv;
  float cx=0.f, cy=0.f, cz=0.f;
  if (i < n){ cx = x[(size_t)i*7+0]-ox; cy = x[(size_t)i*7+1]-oy; cz = x[(size_t)i*7+2]-oz; }
  float b0=3.4e38f,b1=3.4e38f,b2=3.4e38f; int i0=0,i1=0,i2=0;
  for (int base = 0; base < n; base += KNN_TILE){
    int cnt = min(KNN_TILE, n - base);
    for (int t = threadIdx.x; t < cnt; t += blockDim.x){
      sp[t*3+0] = xp[(size_t)(base+t)*7+0];
      sp[t*3+1] = xp[(size_t)(base+t)*7+1];
      sp[t*3+2] = xp[(size_t)(base+t)*7+2];
    }
    __syncthreads();
    if (i < n){
      for (int j = 0; j < cnt; ++j){
        float dx = cx - sp[j*3+0], dy = cy - sp[j*3+1], dz = cz - sp[j*3+2];
        float d = dx*dx + dy*dy + dz*dz;
        int jj = base + j;
        if (d < b2){
          if (d < b0){ b2=b1;i2=i1; b1=b0;i1=i0; b0=d;i0=jj; }
          else if (d < b1){ b2=b1;i2=i1; b1=d;i1=jj; }
          else { b2=d; i2=jj; }
        }
      }
    }
    __syncthreads();
  }
  if (i < n){ idx[i*3+0]=i0; idx[i*3+1]=i1; idx[i*3+2]=i2; }
}
__global__ void k_cf(const float* nf, const float* nft, const int* idx, int n, float* cf){
  int t = blockIdx.x*blockDim.x + threadIdx.x;
  if (t >= n*3) return;
  int i = t / 3;
  int j = idx[t];
  for (int c = 0; c < 11; ++c)
    cf[(size_t)t*11+c] = nf[(size_t)i*11+c] - nft[(size_t)j*11+c];
}
__global__ void k_best(const float* sim, const int* idx, const float* nf, const float* nft,
                       int n, float* diff){
  int i = blockIdx.x*blockDim.x + threadIdx.x;
  if (i >= n) return;
  float s0 = sim[i*3+0], s1 = sim[i*3+1], s2 = sim[i*3+2];
  int b = 0; float sb = s0;
  if (s1 > sb){ b = 1; sb = s1; }
  if (s2 > sb){ b = 2; }
  int j = idx[i*3+b];
  for (int c = 0; c < 11; ++c)
    diff[(size_t)i*11+c] = nf[(size_t)i*11+c] - nft[(size_t)j*11+c];
}
__global__ void k_deg(const int* ei, int E_, float* deg){
  int e = blockIdx.x*blockDim.x + threadIdx.x;
  if (e >= E_) return;
  atomicAdd(&deg[ei[E_+e]], 1.0f);
}
__global__ void k_gcn_init(const float* h, const float* deg, const float* b, int n, int c, float* out){
  int t = blockIdx.x*blockDim.x + threadIdx.x;
  if (t >= n*c) return;
  int i = t / c, j = t % c;
  out[t] = h[t] / (deg[i] + 1.0f) + b[j];     // di*di = 1/(deg+1)
}
__global__ void k_gcn_edge(const float* h, const float* deg, const int* ei, int E_, int c, float* out){
  int e = blockIdx.x*blockDim.x + threadIdx.x;
  if (e >= E_) return;
  int s = ei[e], d = ei[E_+e];
  float w = rsqrtf(deg[s]+1.0f) * rsqrtf(deg[d]+1.0f);
  for (int j = 0; j < c; ++j)
    atomicAdd(&out[(size_t)d*c+j], w * h[(size_t)s*c+j]);
}
__global__ void k_gate(float* nf, const float* mo, int n){
  int t = blockIdx.x*blockDim.x + threadIdx.x;
  if (t >= n) return;
  nf[t] *= sigmoidf_(mo[t]);
}

// ---------------------------------------------------------------------------
// Host orchestration.
// Input flattening assumption: JAX pytree order (dict keys sorted at every
// level, lists/tuples in order):
//   0 edge_index, 1 edge_index_prev, 2..127 params leaves, 128 x, 129 x_prev
// ---------------------------------------------------------------------------
enum {
  IN_EI = 0, IN_EIP = 1,
  P_GCN_W = 2, P_GCN_B = 3,
  P_BN11_B = 4,  P_BN11T_B = 6,  P_BN12_B = 8,  P_BN12T_B = 10,
  P_BN13_B = 12, P_BN13T_B = 14, P_BN14_B = 16, P_BN14T_B = 18,
  P_C11 = 20, P_C11T = 26, P_C12 = 32, P_C12T = 38,   // W,We,a_dst,a_e,a_src,b
  P_FS = 44,                                          // W0,b0,W1,b1
  P_LIN = 48,                                         // W,b
  P_M11 = 50, P_M11T = 56, P_M12 = 62, P_M12T = 68,
  P_M13 = 74, P_M13T = 80, P_M14 = 86, P_M14T = 92,
  P_M17 = 98, P_M17T = 104, P_M18 = 110, P_M18T = 116, P_M19 = 122,
  IN_X = 128, IN_XP = 129
};

extern "C" void kernel_launch(void* const* d_in, const int* in_sizes, int n_in,
                              void* d_out, int out_size, void* d_ws, size_t ws_size,
                              hipStream_t stream)
{
  (void)in_sizes; (void)n_in; (void)out_size; (void)ws_size;
  const int N = 15000, E = 240000;
  const int* EI  = (const int*)d_in[IN_EI];
  const int* EIP = (const int*)d_in[IN_EIP];
  const float* X  = (const float*)d_in[IN_X];
  const float* XP = (const float*)d_in[IN_XP];
  auto P = [&](int i)->const float* { return (const float*)d_in[i]; };

  // ---- workspace layout (~52 MB of f32) ----
  float* w = (float*)d_ws;
  size_t off = 0;
  auto A = [&](size_t nel)->float* { float* p = w + off; off += nel; return p; };
  float* NF   = A((size_t)N*11);    // current-frame node features (persistent)
  float* NFT  = A((size_t)N*16);    // prev-frame node features
  float* NA   = A((size_t)N*64);
  float* NB   = A((size_t)N*64);
  float* NC   = A((size_t)N*64);
  float* ND   = A((size_t)N*64);
  float* EA   = A((size_t)E*16);
  float* EB   = A((size_t)E*16);
  float* Ee   = A((size_t)E);       // attention logits
  float* Ex   = A((size_t)E);       // exp(logit - max)
  float* CFb  = A((size_t)N*3*11);
  float* SIMb = A((size_t)N*3);
  int*   IDXb = (int*)A((size_t)N*3);
  float* STAT = A(136);

  auto cdiv = [](int a, int b){ return (a + b - 1) / b; };
  auto GB   = [&](int M){ return cdiv(cdiv(M,16), WV); };
  const int gN = cdiv(N, TPB), gE = cdiv(E, TPB);
  const float* NUL = nullptr;

  auto bnrun = [&](const float* Xin, int C, int bb, float* Yout){
    k_fill<<<1,128,0,stream>>>(STAT, 0.f, 128);
    k_colstats<<<gN,TPB,0,stream>>>(Xin, N, C, STAT);
    k_bn<<<cdiv(N*C,TPB),TPB,0,stream>>>(Xin, STAT, P(bb+1)/*g*/, P(bb)/*b*/, N, C, Yout);
  };
  // GAT softmax-aggregate; scratch convention: Ee/Ex, m=ND, den=NC
  auto gatrun = [&](const float* h, const float* eh, int o, const int* ei, int g0, float* out){
    k_gat_logits<<<gE,TPB,0,stream>>>(h, eh, P(g0+4)/*a_src*/, P(g0+2)/*a_dst*/, P(g0+3)/*a_e*/,
                                      ei, E, o, Ee);
    k_fill<<<gN,TPB,0,stream>>>(ND, -INFINITY, N);
    k_segmax<<<gE,TPB,0,stream>>>(Ee, ei, E, ND);
    k_fill<<<gN,TPB,0,stream>>>(NC, 0.f, N);
    k_gat_exp<<<gE,TPB,0,stream>>>(Ee, ND, ei, E, Ex, NC);
    k_rowbc<<<cdiv(N*o,TPB),TPB,0,stream>>>(out, P(g0+5)/*b*/, N, o);
    k_gat_agg<<<gE,TPB,0,stream>>>(Ex, NC, h, ei, E, o, out);
  };
  auto stage1 = [&](const float* Xn, const int* ei, int m11, int m14, int g0,
                    int bn1, int m12, int bn2, float* OUT){
    k_mlp<3,64,64,3,3><<<GB(N),TPB,0,stream>>>(Xn, N, 7,
        P(m11),P(m11+1),P(m11+2),P(m11+3),P(m11+4),P(m11+5), 0, NA);          // r
    k_edge_cat<<<gE,TPB,0,stream>>>(NA, ei, E, Xn, 7, 3, 1, EA);              // [E,4]
    k_mlp<4,64,64,7,3><<<GB(E),TPB,0,stream>>>(EA, E, 4,
        P(m14),P(m14+1),P(m14+2),P(m14+3),P(m14+4),P(m14+5), 1, EB);          // edge attr
    k_mlp<7,7,0,0,1><<<GB(N),TPB,0,stream>>>(Xn, N, 7,
        P(g0+0),NUL, NUL,NUL, NUL,NUL, 0, NB);                                // h = x@W
    k_mlp<7,7,0,0,1><<<GB(E),TPB,0,stream>>>(EB, E, 7,
        P(g0+1),NUL, NUL,NUL, NUL,NUL, 0, EA);                                // eh = r4@We
    gatrun(NB, EA, 7, ei, g0, NA);
    bnrun(NA, 7, bn1, NB);
    k_mlp<7,64,64,11,3><<<GB(N),TPB,0,stream>>>(NB, N, 7,
        P(m12),P(m12+1),P(m12+2),P(m12+3),P(m12+4),P(m12+5), 1, NA);
    bnrun(NA, 11, bn2, OUT);
  };
  auto reg2 = [&](const float* Xn, const int* ei, const float* nfb, int m17, int m18){
    k_mlp<3,32,32,3,3><<<GB(N),TPB,0,stream>>>(Xn, N, 7,
        P(m17),P(m17+1),P(m17+2),P(m17+3),P(m17+4),P(m17+5), 0, NA);          // r
    k_edge_cat<<<gE,TPB,0,stream>>>(NA, ei, E, nfb, 11, 0, 11, EA);           // [E,14]
    k_mlp<14,32,32,7,3><<<GB(E),TPB,0,stream>>>(EA, E, 14,
        P(m18),P(m18+1),P(m18+2),P(m18+3),P(m18+4),P(m18+5), 1, EB);          // r2 -> EB
  };
  auto conv2 = [&](const float* nfb, const int* ei, int g0, int bnb, float* outb){
    k_mlp<11,16,0,0,1><<<GB(N),TPB,0,stream>>>(nfb, N, 11,
        P(g0+0),NUL, NUL,NUL, NUL,NUL, 0, NB);                                // h [N,16]
    k_mlp<7,16,0,0,1><<<GB(E),TPB,0,stream>>>(EB, E, 7,
        P(g0+1),NUL, NUL,NUL, NUL,NUL, 0, EA);                                // eh [E,16]
    gatrun(NB, EA, 16, ei, g0, NA);
    bnrun(NA, 16, bnb, outb);
  };

  // ================= forward =================
  stage1(X,  EI,  P_M11,  P_M14,  P_C11,  P_BN11_B,  P_M12,  P_BN12_B,  NF);
  stage1(XP, EIP, P_M11T, P_M14T, P_C11T, P_BN11T_B, P_M12T, P_BN12T_B, NFT);

  // prev-frame second hop
  reg2(XP, EIP, NFT, P_M17T, P_M18T);
  conv2(NFT, EIP, P_C12T, P_BN13T_B, NB);                 // [N,16]
  k_mlp<16,64,64,11,3><<<GB(N),TPB,0,stream>>>(NB, N, 16,
      P(P_M13T),P(P_M13T+1),P(P_M13T+2),P(P_M13T+3),P(P_M13T+4),P(P_M13T+5), 1, NA);
  bnrun(NA, 11, P_BN14T_B, NFT);                          // NFT now [N,11]

  // centroid-corrected brute-force KNN (K=3)
  k_fill<<<1,64,0,stream>>>(STAT, 0.f, 8);
  k_colsum6<<<gN,TPB,0,stream>>>(X, XP, N, STAT);
  k_knn<<<cdiv(N,256),256,0,stream>>>(X, XP, STAT, N, IDXb);
  k_cf<<<cdiv(N*3,TPB),TPB,0,stream>>>(NF, NFT, IDXb, N, CFb);
  k_mlp<11,32,1,0,2><<<GB(N*3),TPB,0,stream>>>(CFb, N*3, 11,
      P(P_FS),P(P_FS+1),P(P_FS+2),P(P_FS+3), NUL,NUL, 0, SIMb);
  k_best<<<gN,TPB,0,stream>>>(SIMb, IDXb, NF, NFT, N, NA); // diff = nf - matched

  // GCN + motion gate
  k_mlp<11,11,0,0,1><<<GB(N),TPB,0,stream>>>(NA, N, 11,
      P(P_GCN_W),NUL, NUL,NUL, NUL,NUL, 0, NB);            // h
  k_fill<<<gN,TPB,0,stream>>>(NC, 0.f, N);
  k_deg<<<gE,TPB,0,stream>>>(EI, E, NC);                   // deg (without +1)
  k_gcn_init<<<cdiv(N*11,TPB),TPB,0,stream>>>(NB, NC, P(P_GCN_B), N, 11, NA);
  k_gcn_edge<<<gE,TPB,0,stream>>>(NB, NC, EI, E, 11, NA);
  k_mlp<11,32,32,11,3><<<GB(N),TPB,0,stream>>>(NA, N, 11,
      P(P_M19),P(P_M19+1),P(P_M19+2),P(P_M19+3),P(P_M19+4),P(P_M19+5), 0, NB); // motion
  k_gate<<<cdiv(N*11,TPB),TPB,0,stream>>>(NF, NB, N*11);   // nf *= sigmoid(motion)

  // current-frame second hop + head
  reg2(X, EI, NF, P_M17, P_M18);
  conv2(NF, EI, P_C12, P_BN13_B, NC);                      // [N,16]
  k_mlp<16,64,64,32,3><<<GB(N),TPB,0,stream>>>(NC, N, 16,
      P(P_M13),P(P_M13+1),P(P_M13+2),P(P_M13+3),P(P_M13+4),P(P_M13+5), 1, NA);
  bnrun(NA, 32, P_BN14_B, NB);                             // [N,32]
  k_mlp<32,1,0,0,1><<<GB(N),TPB,0,stream>>>(NB, N, 32,
      P(P_LIN),P(P_LIN+1), NUL,NUL, NUL,NUL, 2, (float*)d_out); // sigmoid head
}
